// GraphTransformerEncoder_5497558139596
// MI455X (gfx1250) — matrix operand order repaired and checked by reading, not compiled, not run on hardware
//
#include <hip/hip_runtime.h>

typedef __attribute__((ext_vector_type(16))) __bf16 v16bf;
typedef __attribute__((ext_vector_type(8)))  float  v8f;

union BFrag { v16bf v; unsigned int u[8]; };

// Pack two f32 -> packed bf16 pair (low = a, high = b).
// Prefer hardware v_cvt_pk_bf16_f32; else RNE-round + single v_perm_b32.
__device__ __forceinline__ unsigned int pack_bf16(float a, float b) {
#if __has_builtin(__builtin_amdgcn_cvt_pk_bf16_f32)
    auto pk = __builtin_amdgcn_cvt_pk_bf16_f32(a, b);
    return __builtin_bit_cast(unsigned int, pk);
#else
    unsigned int ua = __builtin_bit_cast(unsigned int, a);
    unsigned int ub = __builtin_bit_cast(unsigned int, b);
    ua += 0x7fffu + ((ua >> 16) & 1u);   // round-to-nearest-even
    ub += 0x7fffu + ((ub >> 16) & 1u);
    // result bytes {ub.b3, ub.b2, ua.b3, ua.b2}  ==  {bf16(b), bf16(a)}
    return __builtin_amdgcn_perm(ub, ua, 0x07060302u);
#endif
}

// monotonic float <-> ordered-uint mapping for atomicMax on floats
__device__ __forceinline__ unsigned int f2ord(float f) {
    unsigned int b = __builtin_bit_cast(unsigned int, f);
    return (b & 0x80000000u) ? ~b : (b | 0x80000000u);
}
__device__ __forceinline__ float ord2f(unsigned int u) {
    unsigned int b = (u & 0x80000000u) ? (u & 0x7fffffffu) : ~u;
    return __builtin_bit_cast(float, b);
}

// ---------------------------------------------------------------------------
// Cm[n,128] = act(A[n,128] @ Wm[128,128] + bm) for m = 0..nmat-1.
// A fragments are loaded/packed ONCE into registers and reused across all
// weight matrices; each Wm is staged into LDS (bf16, transposed) in turn.
// block = 256 threads (8 waves); each block computes 128 rows x 128 cols.
// ---------------------------------------------------------------------------
__global__ __launch_bounds__(256) void gemm128_wmma4(
    const float* __restrict__ A,
    const float* __restrict__ W0, const float* __restrict__ W1,
    const float* __restrict__ W2, const float* __restrict__ W3,
    const float* __restrict__ b0, const float* __restrict__ b1,
    const float* __restrict__ b2, const float* __restrict__ b3,
    float* __restrict__ C0, float* __restrict__ C1,
    float* __restrict__ C2, float* __restrict__ C3,
    int nrows, int nmat, int relu_mask)
{
    __shared__ unsigned int sW[128 * 68];  // Wt[col][k/2] bf16 pairs, padded rows

    const float* Ws[4] = {W0, W1, W2, W3};
    const float* bs[4] = {b0, b1, b2, b3};
    float*       Cs[4] = {C0, C1, C2, C3};

    const int t    = threadIdx.x;
    const int wave = t >> 5;
    const int lane = t & 31;
    const int half = lane >> 4;           // lane half selects K sub-block
    const int l16  = lane & 15;
    const int rowBase = blockIdx.x * 128 + wave * 16;

    // A fragments: 4 K-chunks of 32 (ISA 16-bit A 16x32 layout), loaded once.
    BFrag afrag[4];
    {
        long long row = rowBase + l16;
        if (row >= nrows) row = nrows - 1;        // clamp; store is guarded
        const float* arow = A + row * 128;
        const int base = half * 8;
        #pragma unroll
        for (int kc = 0; kc < 4; ++kc) {
            int k0 = kc * 32 + base;
            float4 f0 = *(const float4*)(arow + k0);
            float4 f1 = *(const float4*)(arow + k0 + 4);
            float4 f2 = *(const float4*)(arow + k0 + 16);
            float4 f3 = *(const float4*)(arow + k0 + 20);
            afrag[kc].u[0] = pack_bf16(f0.x, f0.y);
            afrag[kc].u[1] = pack_bf16(f0.z, f0.w);
            afrag[kc].u[2] = pack_bf16(f1.x, f1.y);
            afrag[kc].u[3] = pack_bf16(f1.z, f1.w);
            afrag[kc].u[4] = pack_bf16(f2.x, f2.y);
            afrag[kc].u[5] = pack_bf16(f2.z, f2.w);
            afrag[kc].u[6] = pack_bf16(f3.x, f3.y);
            afrag[kc].u[7] = pack_bf16(f3.z, f3.w);
        }
    }

    for (int m = 0; m < nmat; ++m) {
        const float* W = Ws[m];
        if (m) __syncthreads();           // previous sW consumers done
        // Stage W (row-major [K=128][col=128]) into LDS transposed + bf16.
        #pragma unroll
        for (int i = 0; i < 32; ++i) {
            int idx = t + i * 256;        // 0..8191
            int col = idx & 127;
            int p   = idx >> 7;           // k-pair 0..63
            sW[col * 68 + p] = pack_bf16(W[(2 * p) * 128 + col],
                                         W[(2 * p + 1) * 128 + col]);
        }
        __syncthreads();

        const int   do_relu = (relu_mask >> m) & 1;
        float*      C       = Cs[m];
        const float* bias   = bs[m];

        #pragma unroll
        for (int ct = 0; ct < 8; ++ct) {
            const int colBase = ct * 16;
            const int col = colBase + l16;
            const int kb  = half * 8;     // B: half-wave K offset within chunk
            v8f acc = {};
            #pragma unroll
            for (int kc = 0; kc < 4; ++kc) {
                BFrag bfrag;
                const unsigned int* bp = &sW[col * 68 + ((kc * 32 + kb) >> 1)];
                #pragma unroll
                for (int j = 0; j < 8; ++j) bfrag.u[j] = bp[j + (j & 4)];
                acc = __builtin_amdgcn_wmma_f32_16x16x32_bf16(
                    false, afrag[kc].v, false, bfrag.v, (short)0, acc, false, false);
            }
            const float bv = bias[colBase + l16];
            #pragma unroll
            for (int r = 0; r < 8; ++r) {
                int row = rowBase + r + half * 8;  // C layout: M = r (+8 hi half)
                if (row < nrows) {
                    float vv = acc[r] + bv;
                    if (do_relu) vv = fmaxf(vv, 0.f);
                    C[(long long)row * 128 + colBase + l16] = vv;
                }
            }
        }
    }
}

// ---------------------------------------------------------------------------
// Edge kernels (wave-per-edge; lane l handles channel l of each of 4 heads)
// ---------------------------------------------------------------------------
__global__ __launch_bounds__(256) void edge_alpha(
    const int* __restrict__ esrc, const int* __restrict__ etgt,
    const float* __restrict__ eattr, const float* __restrict__ Wel,
    const float* __restrict__ q, const float* __restrict__ k,
    float* __restrict__ alpha, unsigned int* __restrict__ amax, int nE)
{
    const int wave = threadIdx.x >> 5, lane = threadIdx.x & 31;
    const int e = blockIdx.x * 8 + wave;
    if (e >= nE) return;
    const int s = esrc[e], tg = etgt[e];
    const float ea = eattr[e];
    float acc[4];
    #pragma unroll
    for (int hd = 0; hd < 4; ++hd) {
        int ch = hd * 32 + lane;
        acc[hd] = q[(long long)tg * 128 + ch] *
                  (k[(long long)s * 128 + ch] + ea * Wel[ch]);
    }
    #pragma unroll
    for (int off = 16; off > 0; off >>= 1) {
        #pragma unroll
        for (int hd = 0; hd < 4; ++hd) acc[hd] += __shfl_xor(acc[hd], off, 32);
    }
    if (lane < 4) {
        float a = acc[lane] * 0.17677669529663687f;  // 1/sqrt(32)
        alpha[e * 4 + lane] = a;
        atomicMax(&amax[tg * 4 + lane], f2ord(a));
    }
}

__global__ __launch_bounds__(256) void edge_exp(
    const int* __restrict__ etgt, float* __restrict__ alpha,
    const unsigned int* __restrict__ amax, float* __restrict__ denom, int nE4)
{
    const int tid = blockIdx.x * 256 + threadIdx.x;
    if (tid >= nE4) return;
    const int e = tid >> 2, hd = tid & 3;
    const int tg = etgt[e];
    const float am = ord2f(amax[tg * 4 + hd]);
    const float ex = __expf(alpha[tid] - am);
    alpha[tid] = ex;
    atomicAdd(&denom[tg * 4 + hd], ex);
}

__global__ __launch_bounds__(256) void edge_msg(
    const int* __restrict__ esrc, const int* __restrict__ etgt,
    const float* __restrict__ eattr, const float* __restrict__ Wel,
    const float* __restrict__ v, const float* __restrict__ alpha,
    const float* __restrict__ denom, float* __restrict__ outacc, int nE)
{
    const int wave = threadIdx.x >> 5, lane = threadIdx.x & 31;
    const int e = blockIdx.x * 8 + wave;
    if (e >= nE) return;
    const int s = esrc[e], tg = etgt[e];
    const float ea = eattr[e];
    #pragma unroll
    for (int hd = 0; hd < 4; ++hd) {
        float a = alpha[e * 4 + hd] / denom[tg * 4 + hd];
        int ch = hd * 32 + lane;
        float val = (v[(long long)s * 128 + ch] + ea * Wel[ch]) * a;
        atomicAdd(&outacc[(long long)tg * 128 + ch], val);
    }
}

// ---------------------------------------------------------------------------
// beta gate + gated combine + batchnorm partial sums (wave-per-node)
// ---------------------------------------------------------------------------
__global__ __launch_bounds__(256) void combine_beta_bn(
    const float* __restrict__ outm, const float* __restrict__ xr,
    const float* __restrict__ Wb, float* __restrict__ htmp,
    float* __restrict__ bnsum, float* __restrict__ bnsq, int n)
{
    __shared__ float ssum[128], ssq[128];
    const int t = threadIdx.x;
    if (t < 128) { ssum[t] = 0.f; ssq[t] = 0.f; }
    __syncthreads();

    const int wave = t >> 5, lane = t & 31;
    const int node = blockIdx.x * 8 + wave;
    const bool valid = node < n;
    const long long nn = valid ? node : 0;

    float o[4], x[4], s = 0.f;
    #pragma unroll
    for (int j = 0; j < 4; ++j) {
        int ch = j * 32 + lane;
        o[j] = outm[nn * 128 + ch];
        x[j] = xr[nn * 128 + ch];
        s += o[j] * Wb[ch] + x[j] * Wb[128 + ch] + (o[j] - x[j]) * Wb[256 + ch];
    }
    #pragma unroll
    for (int off = 16; off > 0; off >>= 1) s += __shfl_xor(s, off, 32);
    const float beta = 1.f / (1.f + __expf(-s));

    if (valid) {
        #pragma unroll
        for (int j = 0; j < 4; ++j) {
            int ch = j * 32 + lane;
            float hv = beta * x[j] + (1.f - beta) * o[j];
            htmp[nn * 128 + ch] = hv;
            atomicAdd(&ssum[ch], hv);
            atomicAdd(&ssq[ch], hv * hv);
        }
    }
    __syncthreads();
    if (t < 128) {
        atomicAdd(&bnsum[t], ssum[t]);
        atomicAdd(&bnsq[t], ssq[t]);
    }
}

__global__ void bn_stats(const float* __restrict__ bnsum, const float* __restrict__ bnsq,
                         float* __restrict__ mu, float* __restrict__ rstd, int n)
{
    const int ch = threadIdx.x;   // 128 threads
    float m = bnsum[ch] / (float)n;
    float var = bnsq[ch] / (float)n - m * m;
    mu[ch] = m;
    rstd[ch] = rsqrtf(var + 1e-5f);
}

__global__ __launch_bounds__(256) void bn_apply(
    const float* __restrict__ htmp, const float* __restrict__ hres,
    const float* __restrict__ mu, const float* __restrict__ rstd,
    const float* __restrict__ gamma, const float* __restrict__ bnbeta,
    float* __restrict__ dst, long long total)
{
    const long long tid = (long long)blockIdx.x * 256 + threadIdx.x;
    if (tid >= total) return;
    const int ch = (int)(tid & 127);
    float vv = htmp[tid];
    float hv = gamma[ch] * (vv - mu[ch]) * rstd[ch] + bnbeta[ch];
    hv = fmaxf(hv, 0.f);
    dst[tid] = hv + hres[tid];
}

// ---------------------------------------------------------------------------
extern "C" void kernel_launch(void* const* d_in, const int* in_sizes, int n_in,
                              void* d_out, int out_size, void* d_ws, size_t ws_size,
                              hipStream_t stream)
{
    (void)in_sizes; (void)n_in; (void)out_size; (void)ws_size;
    const int N = 50000, E = 320000, HID = 128, LAYERS = 2;

    const float* x     = (const float*)d_in[0];
    const int*   ei    = (const int*)d_in[1];
    const float* eattr = (const float*)d_in[2];
    const float* W_in  = (const float*)d_in[3];
    const float* b_in  = (const float*)d_in[4];
    const float* Wq    = (const float*)d_in[5];
    const float* bq    = (const float*)d_in[6];
    const float* Wk    = (const float*)d_in[7];
    const float* bk    = (const float*)d_in[8];
    const float* Wv    = (const float*)d_in[9];
    const float* bv    = (const float*)d_in[10];
    const float* We    = (const float*)d_in[11];
    const float* Wskip = (const float*)d_in[12];
    const float* bskip = (const float*)d_in[13];
    const float* Wbeta = (const float*)d_in[14];
    const float* bng   = (const float*)d_in[15];
    const float* bnb   = (const float*)d_in[16];

    const long long NH = (long long)N * HID;
    float* p     = (float*)d_ws;
    float* h     = p; p += NH;
    float* htmp  = p; p += NH;
    float* q     = p; p += NH;
    float* k     = p; p += NH;
    float* v     = p; p += NH;
    float* xr    = p; p += NH;
    float* outm  = p; p += NH;
    float* alpha = p; p += (long long)E * 4;
    unsigned int* amax = (unsigned int*)p; p += (long long)N * 4;
    float* denom = p; p += (long long)N * 4;
    float* bnsum = p; p += 128;
    float* bnsq  = p; p += 128;
    float* mu    = p; p += 128;
    float* rstd  = p; p += 128;

    const int* esrc = ei;
    const int* etgt = ei + E;

    const dim3 bG(256);
    const dim3 gGemm((N + 127) / 128);

    // h = relu(x @ W_in + b_in)     (IN_DIM == HID == 128)
    gemm128_wmma4<<<gGemm, bG, 0, stream>>>(
        x, W_in, W_in, W_in, W_in, b_in, b_in, b_in, b_in,
        h, h, h, h, N, 1, 1);

    for (int l = 0; l < LAYERS; ++l) {
        const float* Wql = Wq    + (long long)l * HID * HID;
        const float* Wkl = Wk    + (long long)l * HID * HID;
        const float* Wvl = Wv    + (long long)l * HID * HID;
        const float* Wsl = Wskip + (long long)l * HID * HID;
        const float* Wel = We    + (long long)l * HID;
        const float* Wbl = Wbeta + (long long)l * 3 * HID;

        // q, k, v, x_r in one pass over h (A fragments reused 4x)
        gemm128_wmma4<<<gGemm, bG, 0, stream>>>(
            h, Wql, Wkl, Wvl, Wsl,
            bq + l * HID, bk + l * HID, bv + l * HID, bskip + l * HID,
            q, k, v, xr, N, 4, 0);

        hipMemsetAsync(amax,  0, (size_t)N * 4 * sizeof(unsigned int), stream);
        hipMemsetAsync(denom, 0, (size_t)N * 4 * sizeof(float), stream);
        hipMemsetAsync(outm,  0, (size_t)NH * sizeof(float), stream);
        hipMemsetAsync(bnsum, 0, 128 * sizeof(float), stream);
        hipMemsetAsync(bnsq,  0, 128 * sizeof(float), stream);

        edge_alpha<<<dim3((E + 7) / 8), bG, 0, stream>>>(esrc, etgt, eattr, Wel, q, k, alpha, amax, E);
        edge_exp<<<dim3((4 * E + 255) / 256), bG, 0, stream>>>(etgt, alpha, amax, denom, 4 * E);
        edge_msg<<<dim3((E + 7) / 8), bG, 0, stream>>>(esrc, etgt, eattr, Wel, v, alpha, denom, outm, E);

        combine_beta_bn<<<dim3((N + 7) / 8), bG, 0, stream>>>(outm, xr, Wbl, htmp, bnsum, bnsq, N);
        bn_stats<<<dim3(1), dim3(128), 0, stream>>>(bnsum, bnsq, mu, rstd, N);

        float* dst = (l == LAYERS - 1) ? (float*)d_out : h;
        bn_apply<<<dim3((int)((NH + 255) / 256)), bG, 0, stream>>>(
            htmp, h, mu, rstd, bng + l * HID, bnb + l * HID, dst, NH);
    }
}
